// BasicBlock_73272142069990
// MI455X (gfx1250) — compile-verified, matrix operand order checked by reading, not run on hardware
//
#include <hip/hip_runtime.h>
#include <hip/hip_bf16.h>

// ---------------------------------------------------------------------------
// Binarized residual block on gfx1250 via V_WMMA_I32_16X16X64_IU8.
// Geometry: N=64, C=256, H=W=28, 3x3 conv pad 1 -> implicit GEMM
//   M = 64*28*28 = 50176, Ncols = 256, K = 256*9 = 2304 (36 k-blocks of 64)
// GEMM main loop: TRIPLE-buffered LDS tiles filled with
// GLOBAL_LOAD_ASYNC_TO_LDS_B128 (ASYNCcnt), one barrier per k-step,
// consumed by WMMA iu8.
// ---------------------------------------------------------------------------

typedef __attribute__((ext_vector_type(8))) int v8i;
typedef int v4i_t __attribute__((__vector_size__(16)));

#define CCH   256
#define HWW   784          // 28*28
#define NIMG  64
#define MROWS 50176        // NIMG*HWW
#define KTOT  2304
#define KBLK  36           // KTOT / 64
#define CNT_F 50176.0f
#define EPSBN 1e-5f

#if defined(__HIP_DEVICE_COMPILE__) && __has_builtin(__builtin_amdgcn_global_load_async_to_lds_b128)
#define HAVE_ASYNC 1
#else
#define HAVE_ASYNC 0
#endif

#define GLOBAL_AS __attribute__((address_space(1)))
#define LDS_AS    __attribute__((address_space(3)))

__device__ __forceinline__ void async_cp16(const void* g, void* l) {
#if HAVE_ASYNC
    // per-lane 16B global -> LDS, tracked by ASYNCcnt
    __builtin_amdgcn_global_load_async_to_lds_b128(
        (GLOBAL_AS v4i_t*)(uintptr_t)g,
        (LDS_AS v4i_t*)(uintptr_t)l, 0, 0);
#else
    *(int4*)l = *(const int4*)g;
#endif
}

__device__ __forceinline__ void wait_async_le4() {
#if HAVE_ASYNC
#if __has_builtin(__builtin_amdgcn_s_wait_asynccnt)
    __builtin_amdgcn_s_wait_asynccnt(4);
#else
    asm volatile("s_wait_asynccnt 0x4" ::: "memory");
#endif
#endif
}

__device__ __forceinline__ void wait_async_le0() {
#if HAVE_ASYNC
#if __has_builtin(__builtin_amdgcn_s_wait_asynccnt)
    __builtin_amdgcn_s_wait_asynccnt(0);
#else
    asm volatile("s_wait_asynccnt 0x0" ::: "memory");
#endif
#endif
}

__device__ __forceinline__ signed char sign8(float v) {
    return v > 0.f ? (signed char)1 : (v < 0.f ? (signed char)-1 : (signed char)0);
}

// ---------------------------------------------------------------------------
// Pack x (NCHW f32) -> sign int8 NHWC; accumulate per-channel sum/sumsq of x.
// ---------------------------------------------------------------------------
__global__ __launch_bounds__(256) void pack_x_kernel(const float* __restrict__ x,
                                                     signed char* __restrict__ sx,
                                                     float* __restrict__ xsum,
                                                     float* __restrict__ xss) {
    const int n = blockIdx.x >> 8;
    const int c = blockIdx.x & 255;
    const float* src = x + ((size_t)(n * CCH + c)) * HWW;
    float s = 0.f, q = 0.f;
    for (int i = threadIdx.x; i < HWW; i += 256) {
        float v = src[i];
        s += v; q += v * v;
        sx[(((size_t)n * HWW + i) << 8) + c] = sign8(v);
    }
    __shared__ float rs[256], rq[256];
    rs[threadIdx.x] = s; rq[threadIdx.x] = q;
    __syncthreads();
    for (int off = 128; off > 0; off >>= 1) {
        if (threadIdx.x < off) {
            rs[threadIdx.x] += rs[threadIdx.x + off];
            rq[threadIdx.x] += rq[threadIdx.x + off];
        }
        __syncthreads();
    }
    if (threadIdx.x == 0) {
        atomicAdd(&xsum[c], rs[0]);
        atomicAdd(&xss[c],  rq[0]);
    }
}

// ---------------------------------------------------------------------------
// Pack weights (OIHW f32) -> sign int8, K-blocked: Bpk[(kb*256 + cout)*64 + j]
// ---------------------------------------------------------------------------
__global__ __launch_bounds__(256) void pack_w_kernel(const float* __restrict__ w,
                                                     signed char* __restrict__ bpk) {
    int o = blockIdx.x * 256 + threadIdx.x;          // 0 .. 589823
    int j      = o & 63;
    int coutkb = o >> 6;
    int cout   = coutkb & 255;
    int kb     = coutkb >> 8;
    int k      = kb * 64 + j;
    int tap    = k >> 8;
    int cin    = k & 255;
    bpk[o] = sign8(w[((size_t)(cout * CCH + cin)) * 9 + tap]);
}

// ---------------------------------------------------------------------------
// Implicit-GEMM binary conv. Block tile 128(M) x 128(N), 8 waves, each 32x64.
// Triple-buffered LDS tiles; async global->LDS staging; WMMA iu8 compute.
// ---------------------------------------------------------------------------
__global__ __launch_bounds__(256) void bgemm_conv_kernel(const signed char* __restrict__ sx,
                                                         const signed char* __restrict__ bpk,
                                                         int* __restrict__ cout_buf,
                                                         const signed char* __restrict__ zpad) {
    __shared__ signed char Atile[3][128 * 64];
    __shared__ signed char Btile[3][128 * 64];

    const int nb = blockIdx.x & 1;           // 2 column blocks of 128
    const int mb = blockIdx.x >> 1;          // 392 row blocks of 128

    const int lane   = threadIdx.x & 31;
    const int wid    = threadIdx.x >> 5;     // 0..7
    const int wave_m = wid & 3;              // 4 waves along M
    const int wave_n = wid >> 2;             // 2 waves along N
    const int mrow   = lane & 15;
    const int khalf  = (lane & 16) >> 1;     // +8 K for lanes 16..31
    const int rowsel = (lane & 16) >> 1;     // +8 M rows for C/D hi lanes

    // ---- hoisted per-thread staging descriptors (2 quads of 16B each) ----
    const signed char* rowptr[2];
    int ohv[2], owv[2];
#pragma unroll
    for (int i = 0; i < 2; ++i) {
        const int q   = threadIdx.x + (i << 8);
        const int r   = q >> 2;
        const int sub = q & 3;
        const int m   = mb * 128 + r;
        const int ni  = m / HWW;
        const int rem = m - ni * HWW;
        const int oh  = rem / 28;
        const int ow  = rem - oh * 28;
        ohv[i] = oh; owv[i] = ow;
        rowptr[i] = sx + ((((size_t)ni * HWW + (size_t)(oh * 28 + ow)) << 8) + (sub << 4));
    }

    auto stage = [&](int buf, int kb) {
        const int tap  = kb >> 2;
        const int cin0 = (kb & 3) << 6;
        const int dy   = tap / 3 - 1;
        const int dx   = tap - (tap / 3) * 3 - 1;
        const int shift = ((dy * 28 + dx) << 8) + cin0;
#pragma unroll
        for (int i = 0; i < 2; ++i) {
            const int q  = threadIdx.x + (i << 8);
            const bool ok = ((unsigned)(ohv[i] + dy) < 28u) & ((unsigned)(owv[i] + dx) < 28u);
            const signed char* g = ok ? (rowptr[i] + shift) : zpad;
            async_cp16(g, &Atile[buf][q << 4]);
        }
        const signed char* bs = bpk + (((size_t)(kb * 256 + nb * 128)) << 6);
#pragma unroll
        for (int i = 0; i < 2; ++i) {
            const int q = threadIdx.x + (i << 8);
            async_cp16(bs + (q << 4), &Btile[buf][q << 4]);
        }
    };

    v8i acc[2][4];
#pragma unroll
    for (int mt = 0; mt < 2; ++mt)
#pragma unroll
        for (int nt = 0; nt < 4; ++nt)
            acc[mt][nt] = v8i{0, 0, 0, 0, 0, 0, 0, 0};

    // ---- fragment load + 8x WMMA from one LDS buffer ----
    auto mma_step = [&](int buf) {
        v8i afrag[2];
#pragma unroll
        for (int mt = 0; mt < 2; ++mt) {
            const signed char* base = &Atile[buf][(((wave_m << 5) + (mt << 4) + mrow) << 6)];
#pragma unroll
            for (int v = 0; v < 8; ++v) {
                const int koff = ((v & 1) << 2) | ((v & 2) << 3) | ((v & 4) << 3) | khalf;
                afrag[mt][v] = *(const int*)(base + koff);
            }
        }
        v8i bfrag[4];
#pragma unroll
        for (int nt = 0; nt < 4; ++nt) {
            const signed char* base = &Btile[buf][(((wave_n << 6) + (nt << 4) + mrow) << 6)];
#pragma unroll
            for (int v = 0; v < 8; ++v) {
                const int koff = ((v & 1) << 2) | ((v & 2) << 3) | ((v & 4) << 3) | khalf;
                bfrag[nt][v] = *(const int*)(base + koff);
            }
        }
#pragma unroll
        for (int mt = 0; mt < 2; ++mt)
#pragma unroll
            for (int nt = 0; nt < 4; ++nt)
                acc[mt][nt] = __builtin_amdgcn_wmma_i32_16x16x64_iu8(
                    /*sgn_a=*/true, afrag[mt],
                    /*sgn_b=*/true, bfrag[nt],
                    acc[mt][nt], /*reuse_a=*/false, /*reuse_b=*/false);
    };

    stage(0, 0);                              // preload first tile

    // K loop unrolled by 3: static buffer indices, one barrier per k-step.
    for (int t = 0; t < KBLK / 3; ++t) {
        const int kb = 3 * t;
        // step 0: compute buf0, prefetch kb+1 -> buf1
        stage(1, kb + 1);
        wait_async_le4();
        __syncthreads();
        mma_step(0);
        // step 1: compute buf1, prefetch kb+2 -> buf2
        stage(2, kb + 2);
        wait_async_le4();
        __syncthreads();
        mma_step(1);
        // step 2: compute buf2, prefetch kb+3 -> buf0
        if (t + 1 < KBLK / 3) {
            stage(0, kb + 3);
            wait_async_le4();
        } else {
            wait_async_le0();
        }
        __syncthreads();
        mma_step(2);
    }

    // ---- store C (int32, NHWC rows of 256) ----
#pragma unroll
    for (int mt = 0; mt < 2; ++mt)
#pragma unroll
        for (int nt = 0; nt < 4; ++nt)
#pragma unroll
            for (int r = 0; r < 8; ++r) {
                const int mloc = (wave_m << 5) + (mt << 4) + r + rowsel;
                const int col  = (nb << 7) + (wave_n << 6) + (nt << 4) + (lane & 15);
                cout_buf[((size_t)(mb * 128 + mloc) << 8) + col] = acc[mt][nt][r];
            }
}

// ---------------------------------------------------------------------------
// Per-channel sum / sumsq over NHWC [MROWS][256] (int32 or f32 source).
// ---------------------------------------------------------------------------
__global__ __launch_bounds__(256) void stats_kernel(const void* __restrict__ in,
                                                    int is_int,
                                                    float* __restrict__ sum,
                                                    float* __restrict__ ss) {
    const int c = threadIdx.x;
    const int row0 = blockIdx.x * 256;
    float s = 0.f, q = 0.f;
    for (int i = 0; i < 256; ++i) {
        const size_t idx = ((size_t)(row0 + i) << 8) + c;
        const float v = is_int ? (float)((const int*)in)[idx]
                               : ((const float*)in)[idx];
        s += v; q += v * v;
    }
    atomicAdd(&sum[c], s);
    atomicAdd(&ss[c],  q);
}

// ---------------------------------------------------------------------------
// Epilogue 1: residual = prelu(bn1(conv1) + bn2(x)); also sign(residual).
// ---------------------------------------------------------------------------
__global__ __launch_bounds__(256) void epi1_kernel(const int* __restrict__ c1,
                                                   const float* __restrict__ x,
                                                   const float* __restrict__ s1, const float* __restrict__ q1,
                                                   const float* __restrict__ s2, const float* __restrict__ q2,
                                                   const float* __restrict__ g1, const float* __restrict__ b1,
                                                   const float* __restrict__ g2, const float* __restrict__ b2,
                                                   const float* __restrict__ a1,
                                                   float* __restrict__ resf,
                                                   signed char* __restrict__ sres) {
    const size_t gid = (size_t)blockIdx.x * 256 + threadIdx.x;
    const int c = threadIdx.x;
    const size_t m = gid >> 8;
    const float inv = 1.0f / CNT_F;

    const float m1 = s1[c] * inv;
    const float v1 = q1[c] * inv - m1 * m1;
    const float bnc = ((float)c1[gid] - m1) * rsqrtf(v1 + EPSBN) * g1[c] + b1[c];

    const int n = (int)(m / HWW);
    const int p = (int)(m - (size_t)n * HWW);
    const float xv = x[((size_t)(n * CCH + c)) * HWW + p];
    const float m2 = s2[c] * inv;
    const float v2 = q2[c] * inv - m2 * m2;
    const float bnx = (xv - m2) * rsqrtf(v2 + EPSBN) * g2[c] + b2[c];

    const float o = bnc + bnx;
    const float r = o >= 0.f ? o : a1[c] * o;
    resf[gid] = r;
    sres[gid] = sign8(r);
}

// ---------------------------------------------------------------------------
// Epilogue 2: out = prelu(bn4(conv2) + bn3(residual)), written NCHW fp32.
// ---------------------------------------------------------------------------
__global__ __launch_bounds__(256) void epi2_kernel(const int* __restrict__ c2,
                                                   const float* __restrict__ resf,
                                                   const float* __restrict__ s4, const float* __restrict__ q4,
                                                   const float* __restrict__ s3, const float* __restrict__ q3,
                                                   const float* __restrict__ g4, const float* __restrict__ b4,
                                                   const float* __restrict__ g3, const float* __restrict__ b3,
                                                   const float* __restrict__ a2,
                                                   float* __restrict__ out) {
    const size_t gid = (size_t)blockIdx.x * 256 + threadIdx.x;
    const int c = threadIdx.x;
    const size_t m = gid >> 8;
    const float inv = 1.0f / CNT_F;

    const float m4 = s4[c] * inv;
    const float v4 = q4[c] * inv - m4 * m4;
    const float bnc = ((float)c2[gid] - m4) * rsqrtf(v4 + EPSBN) * g4[c] + b4[c];

    const float rv = resf[gid];
    const float m3 = s3[c] * inv;
    const float v3 = q3[c] * inv - m3 * m3;
    const float bnr = (rv - m3) * rsqrtf(v3 + EPSBN) * g3[c] + b3[c];

    const float o = bnc + bnr;
    const float r = o >= 0.f ? o : a2[c] * o;

    const int n = (int)(m / HWW);
    const int p = (int)(m - (size_t)n * HWW);
    out[((size_t)(n * CCH + c)) * HWW + p] = r;
}

// ---------------------------------------------------------------------------
// Host orchestration
// ---------------------------------------------------------------------------
extern "C" void kernel_launch(void* const* d_in, const int* in_sizes, int n_in,
                              void* d_out, int out_size, void* d_ws, size_t ws_size,
                              hipStream_t stream) {
    (void)in_sizes; (void)n_in; (void)out_size; (void)ws_size;

    const float* x  = (const float*)d_in[0];
    const float* w1 = (const float*)d_in[1];
    const float* w2 = (const float*)d_in[2];
    const float* g1 = (const float*)d_in[3];
    const float* b1 = (const float*)d_in[4];
    const float* g2 = (const float*)d_in[5];
    const float* b2 = (const float*)d_in[6];
    const float* g3 = (const float*)d_in[7];
    const float* b3 = (const float*)d_in[8];
    const float* g4 = (const float*)d_in[9];
    const float* b4 = (const float*)d_in[10];
    const float* a1 = (const float*)d_in[11];
    const float* a2 = (const float*)d_in[12];
    float* out = (float*)d_out;

    // ---- workspace carve-up ----
    char* ws = (char*)d_ws;
    float* stats = (float*)ws;                    // 8 arrays x 256 floats
    float* xsum  = stats + 0 * 256;
    float* xss   = stats + 1 * 256;
    float* c1sum = stats + 2 * 256;
    float* c1ss  = stats + 3 * 256;
    float* rsum  = stats + 4 * 256;
    float* rss   = stats + 5 * 256;
    float* c2sum = stats + 6 * 256;
    float* c2ss  = stats + 7 * 256;

    signed char* zpad = (signed char*)(ws + 8192);         // 256 B of zeros

    const size_t SX_BYTES  = (size_t)MROWS * 256;          // 12.85 MB int8
    const size_t BPK_BYTES = (size_t)KBLK * 256 * 64;      // 0.59 MB int8
    const size_t C_BYTES   = (size_t)MROWS * 256 * 4;      // 51.4 MB

    signed char* sx   = (signed char*)(ws + 8192 + 256);
    signed char* sres = sx + SX_BYTES;
    signed char* bpk1 = sres + SX_BYTES;
    signed char* bpk2 = bpk1 + BPK_BYTES;
    int*   c1    = (int*)(bpk2 + BPK_BYTES);
    float* resf  = (float*)c1;                              // aliases c1 (in-place)
    int*   c2    = (int*)((char*)c1 + C_BYTES);

    (void)hipMemsetAsync(ws, 0, 8192 + 256, stream);        // stats + zero pad

    // 1) pack activations + bn2 stats
    pack_x_kernel<<<NIMG * CCH, 256, 0, stream>>>(x, sx, xsum, xss);
    // 2) pack both weight tensors
    pack_w_kernel<<<KBLK * 256 * 64 / 256, 256, 0, stream>>>(w1, bpk1);
    pack_w_kernel<<<KBLK * 256 * 64 / 256, 256, 0, stream>>>(w2, bpk2);
    // 3) binary conv1 (WMMA iu8, async LDS staging)
    bgemm_conv_kernel<<<(MROWS / 128) * 2, 256, 0, stream>>>(sx, bpk1, c1, zpad);
    // 4) bn1 stats over conv1
    stats_kernel<<<MROWS / 256, 256, 0, stream>>>(c1, 1, c1sum, c1ss);
    // 5) fused bn1 + bn2 + add + prelu -> residual (+signs)
    epi1_kernel<<<MROWS, 256, 0, stream>>>(c1, x, c1sum, c1ss, xsum, xss,
                                           g1, b1, g2, b2, a1, resf, sres);
    // 6) bn3 stats over residual
    stats_kernel<<<MROWS / 256, 256, 0, stream>>>(resf, 0, rsum, rss);
    // 7) binary conv2
    bgemm_conv_kernel<<<(MROWS / 128) * 2, 256, 0, stream>>>(sres, bpk2, c2, zpad);
    // 8) bn4 stats over conv2
    stats_kernel<<<MROWS / 256, 256, 0, stream>>>(c2, 1, c2sum, c2ss);
    // 9) fused bn4 + bn3 + add + prelu -> NCHW output
    epi2_kernel<<<MROWS, 256, 0, stream>>>(c2, resf, c2sum, c2ss, rsum, rss,
                                           g4, b4, g3, b3, a2, out);
}
